// RecurrentDecoderConvLSTM_3736621547809
// MI455X (gfx1250) — matrix operand order repaired and checked by reading, not compiled, not run on hardware
//
#include <hip/hip_runtime.h>
#include <hip/hip_bf16.h>

typedef __bf16 bf16;
typedef __attribute__((ext_vector_type(16))) __bf16 v16bf;
typedef __attribute__((ext_vector_type(8)))  float  v8f;

#define NWG_LSTM 32u

// ---------------------------------------------------------------------------
// WMMA helpers (CDNA5 wave32, 16x16x32 bf16 -> f32)
// ---------------------------------------------------------------------------
static __device__ __forceinline__ v8f wmma_bf16(v16bf a, v16bf b, v8f c) {
  return __builtin_amdgcn_wmma_f32_16x16x32_bf16(false, a, false, b, (short)0, c,
                                                 false, false);
}

// A fragment: rows row0..row0+15, K cols k0..k0+31 of row-major bf16 [M x lda].
// Per ISA 16-bit A layout: lane<16 holds K {0..7,16..23}, lane>=16 holds {8..15,24..31}.
static __device__ __forceinline__ v16bf load_a16(const bf16* __restrict__ A, int lda,
                                                 int row0, int k0) {
  int lane = threadIdx.x & 31;
  int m = lane & 15;
  int kb = (lane & 16) ? 8 : 0;
  const bf16* p = A + (size_t)(row0 + m) * lda + k0 + kb;
  v16bf a;
#pragma unroll
  for (int j = 0; j < 8; ++j) { a[j] = p[j]; a[j + 8] = p[j + 16]; }
  return a;
}

// A fragment with batchnorm applied on load: a = x*scale[k] + shift[k]
static __device__ __forceinline__ v16bf load_a16_bn(const bf16* __restrict__ A, int lda,
                                                    int row0, int k0,
                                                    const float* __restrict__ scale,
                                                    const float* __restrict__ shift) {
  int lane = threadIdx.x & 31;
  int m = lane & 15;
  int kb = (lane & 16) ? 8 : 0;
  const bf16* p = A + (size_t)(row0 + m) * lda + k0 + kb;
  const float* sc = scale + k0 + kb;
  const float* sh = shift + k0 + kb;
  v16bf a;
#pragma unroll
  for (int j = 0; j < 8; ++j) {
    a[j]     = (bf16)((float)p[j]      * sc[j]      + sh[j]);
    a[j + 8] = (bf16)((float)p[j + 16] * sc[j + 16] + sh[j + 16]);
  }
  return a;
}

// B fragment from W stored row-major [N x ldw] representing B = W^T
// (B[k][n] = W[n][k]).  Lane<16 holds K {0..7,16..23}, lane>=16 holds {8..15,24..31}.
static __device__ __forceinline__ v16bf load_bT16(const bf16* __restrict__ W, int ldw,
                                                  int col0, int k0) {
  int lane = threadIdx.x & 31;
  int n = lane & 15;
  int kb = (lane & 16) ? 8 : 0;
  const bf16* p = W + (size_t)(col0 + n) * ldw + k0 + kb;
  v16bf b;
#pragma unroll
  for (int j = 0; j < 8; ++j) { b[j] = p[j]; b[j + 8] = p[j + 16]; }
  return b;
}

static __device__ __forceinline__ float sigmoidf_(float x) {
  return 1.0f / (1.0f + __expf(-x));
}

// ---------------------------------------------------------------------------
// Grid-wide sense barrier (32 WGs, persistent LSTM kernel)
// ---------------------------------------------------------------------------
static __device__ __forceinline__ void grid_sync(unsigned* cnt, unsigned target) {
  __syncthreads();
  if (threadIdx.x == 0) {
    __builtin_amdgcn_fence(__ATOMIC_RELEASE, "agent");   // global_wb
    atomicAdd(cnt, 1u);
    while (__hip_atomic_load(cnt, __ATOMIC_RELAXED, __HIP_MEMORY_SCOPE_AGENT) < target)
      __builtin_amdgcn_s_sleep(2);
    __builtin_amdgcn_fence(__ATOMIC_ACQUIRE, "agent");   // global_inv
  }
  __syncthreads();
}

// ---------------------------------------------------------------------------
// Prep kernel: bf16 conversions + transposes + accumulator/counter reset
// ---------------------------------------------------------------------------
__global__ void __launch_bounds__(256) prep_kernel(
    const float* __restrict__ h0, const float* __restrict__ w_dc1,
    const float* __restrict__ w_dc2, bf16* __restrict__ hA,
    bf16* __restrict__ wd1T, bf16* __restrict__ wd2T,
    float* __restrict__ bnsum, float* __restrict__ bnsq,
    unsigned* __restrict__ sync_cnt) {
  long idx = (long)blockIdx.x * 256 + threadIdx.x;
  if (idx < 32768) { hA[idx] = (bf16)h0[idx]; return; }           // h0 -> bf16
  idx -= 32768;
  if (idx < 196608) {                                             // wd1T[k][co][ci]
    int k = (int)(idx / 65536), r = (int)(idx % 65536);
    int co = r >> 8, ci = r & 255;
    wd1T[idx] = (bf16)w_dc1[(size_t)(ci * 256 + co) * 3 + k];
    return;
  }
  idx -= 196608;
  if (idx < 196608) {                                             // wd2T[k][co][ci]
    int k = (int)(idx / 65536), r = (int)(idx % 65536);
    int co = r >> 8, ci = r & 255;
    wd2T[idx] = (bf16)w_dc2[(size_t)(ci * 256 + co) * 3 + k];
    return;
  }
  idx -= 196608;
  if (idx < 256) { bnsum[idx] = 0.f; return; }
  idx -= 256;
  if (idx < 256) { bnsq[idx] = 0.f; return; }
  idx -= 256;
  if (idx == 0) *sync_cnt = 0u;
}

// ---------------------------------------------------------------------------
// Persistent LSTM scan kernel: 32 WGs x 256 threads, weights in LDS
//   WG g owns hidden units [g*16, g*16+16) across all 4 gates.
// ---------------------------------------------------------------------------
static __device__ void dense_phase(const bf16* __restrict__ h, bf16* __restrict__ xout,
                                   const bf16* sWd, float* pbuf, const float* sBd,
                                   int g) {
  const int tid = threadIdx.x, wave = tid >> 5, lane = tid & 31;
  const int ct = g >> 1;
  const int s = wave >> 2, ch = wave & 3;
  const int rt = (g & 1) * 2 + s;
  const int kbase = ch * 128;
  v8f acc = {0.f, 0.f, 0.f, 0.f, 0.f, 0.f, 0.f, 0.f};
#pragma unroll
  for (int i = 0; i < 4; ++i) {
    v16bf a = load_a16(h, 512, rt * 16, kbase + i * 32);
    v16bf b = load_bT16(sWd, 512, 0, kbase + i * 32);
    acc = wmma_bf16(a, b, acc);
  }
#pragma unroll
  for (int v = 0; v < 8; ++v)
    pbuf[((s * 4 + ch) * 32 + lane) * 8 + v] = acc[v];
  __syncthreads();
  for (int o = tid; o < 512; o += 256) {   // reduce K-partials, add bias, emit bf16
    int ss = o >> 8, rem = o & 255, mm = rem >> 4, n = rem & 15;
    int rrt = (g & 1) * 2 + ss;
    int ln = ((mm >> 3) << 4) | n, v = mm & 7;
    float val = sBd[n];
#pragma unroll
    for (int c = 0; c < 4; ++c) val += pbuf[((ss * 4 + c) * 32 + ln) * 8 + v];
    xout[(size_t)(rrt * 16 + mm) * 256 + ct * 16 + n] = (bf16)val;
  }
}

// Wave w owns row-tile rt = w>>1 and gate pair q0 = (w&1)*2 : A fragment shared
// between the two gate tiles -> one A load feeds two WMMAs (dual accumulators).
static __device__ void gates_phase(const bf16* __restrict__ x, const bf16* __restrict__ h,
                                   const bf16* sWih, const bf16* sWhh, float* gbuf,
                                   const float* sB) {
  const int tid = threadIdx.x, wave = tid >> 5, lane = tid & 31;
  const int rt = wave >> 1;
  const int q0 = (wave & 1) * 2;
  v8f acc0 = {0.f, 0.f, 0.f, 0.f, 0.f, 0.f, 0.f, 0.f};
  v8f acc1 = {0.f, 0.f, 0.f, 0.f, 0.f, 0.f, 0.f, 0.f};
  for (int k0 = 0; k0 < 256; k0 += 32) {      // x @ W_ih^T slice
    v16bf a = load_a16(x, 256, rt * 16, k0);
    acc0 = wmma_bf16(a, load_bT16(sWih, 256, q0 * 16, k0), acc0);
    acc1 = wmma_bf16(a, load_bT16(sWih, 256, (q0 + 1) * 16, k0), acc1);
  }
  for (int k0 = 0; k0 < 512; k0 += 32) {      // h @ W_hh^T slice
    v16bf a = load_a16(h, 512, rt * 16, k0);
    acc0 = wmma_bf16(a, load_bT16(sWhh, 512, q0 * 16, k0), acc0);
    acc1 = wmma_bf16(a, load_bT16(sWhh, 512, (q0 + 1) * 16, k0), acc1);
  }
  const int n = lane & 15;
#pragma unroll
  for (int s = 0; s < 2; ++s) {
    const int q = q0 + s;
    const v8f acc = s ? acc1 : acc0;
    const float bias = sB[q * 16 + n];
#pragma unroll
    for (int v = 0; v < 8; ++v) {
      int rr = ((lane >> 4) << 3) + v;
      float d = acc[v] + bias;
      d = (q == 2) ? tanhf(d) : sigmoidf_(d);
      gbuf[(rt * 16 + rr) * 64 + q * 16 + n] = d;
    }
  }
  __syncthreads();
}

static __device__ void update_phase(bf16* __restrict__ hout, const float* gbuf,
                                    float* cbuf, int g) {
  for (int idx = threadIdx.x; idx < 1024; idx += 256) {
    int b = idx >> 4, u = idx & 15;
    float iv = gbuf[b * 64 + u];
    float fv = gbuf[b * 64 + 16 + u];
    float gv = gbuf[b * 64 + 32 + u];
    float ov = gbuf[b * 64 + 48 + u];
    float c2 = fv * cbuf[idx] + iv * gv;
    cbuf[idx] = c2;
    hout[(size_t)b * 512 + g * 16 + u] = (bf16)(ov * tanhf(c2));
  }
}

__global__ void __launch_bounds__(256, 1) lstm_kernel(
    const float* __restrict__ W_ih, const float* __restrict__ b_ih,
    const float* __restrict__ W_hh, const float* __restrict__ b_hh,
    const float* __restrict__ W_dense, const float* __restrict__ b_dense,
    const float* __restrict__ c0, bf16* __restrict__ hA, bf16* __restrict__ hB,
    bf16* __restrict__ xbuf, unsigned* __restrict__ sync_cnt) {
  extern __shared__ char smem[];
  bf16* sWih = (bf16*)smem;                    // 64 x 256
  bf16* sWhh = sWih + 64 * 256;                // 64 x 512
  bf16* sWd  = sWhh + 64 * 512;                // 16 x 512
  float* gbuf = (float*)(sWd + 16 * 512);      // 64 x 64 (reused as dense pbuf)
  float* cbuf = gbuf + 64 * 64;                // 64 x 16 persistent cell state
  float* sB   = cbuf + 64 * 16;                // 64 fused gate biases
  float* sBd  = sB + 64;                       // 16 dense biases

  const int g = blockIdx.x;                    // 0..31
  const int tid = threadIdx.x;
  const int ct = g >> 1;

  // ---- preload weight slices (f32 -> bf16) into LDS ----
  for (int idx = tid; idx < 64 * 256; idx += 256) {
    int lc = idx >> 8, k = idx & 255;
    int grow = (lc >> 4) * 512 + g * 16 + (lc & 15);
    sWih[idx] = (bf16)W_ih[(size_t)grow * 256 + k];
  }
  for (int idx = tid; idx < 64 * 512; idx += 256) {
    int lc = idx >> 9, k = idx & 511;
    int grow = (lc >> 4) * 512 + g * 16 + (lc & 15);
    sWhh[idx] = (bf16)W_hh[(size_t)grow * 512 + k];
  }
  for (int idx = tid; idx < 16 * 512; idx += 256) {
    int lc = idx >> 9, k = idx & 511;
    sWd[idx] = (bf16)W_dense[(size_t)(ct * 16 + lc) * 512 + k];
  }
  for (int idx = tid; idx < 64 * 16; idx += 256) {
    int b = idx >> 4, u = idx & 15;
    cbuf[idx] = c0[(size_t)b * 512 + g * 16 + u];
  }
  for (int idx = tid; idx < 64; idx += 256) {
    int grow = (idx >> 4) * 512 + g * 16 + (idx & 15);
    sB[idx] = b_ih[grow] + b_hh[grow];
  }
  if (tid < 16) sBd[tid] = b_dense[ct * 16 + tid];
  __syncthreads();

  unsigned gen = 0;
  // x0 = dense(h0)
  dense_phase(hA, xbuf, sWd, gbuf, sBd, g);
  grid_sync(sync_cnt, NWG_LSTM * (++gen));

  for (int t = 0; t < 512; ++t) {
    const bf16* hin = (t & 1) ? hB : hA;
    bf16* hout = (t & 1) ? hA : hB;
    gates_phase(xbuf + (size_t)t * 16384, hin, sWih, sWhh, gbuf, sB);
    update_phase(hout, gbuf, cbuf, g);
    grid_sync(sync_cnt, NWG_LSTM * (++gen));
    dense_phase(hout, xbuf + (size_t)(t + 1) * 16384, sWd, gbuf, sBd, g);
    grid_sync(sync_cnt, NWG_LSTM * (++gen));
  }
}

// ---------------------------------------------------------------------------
// Deconv1 (stride-2 transposed conv as per-parity GEMMs) + ReLU + BN partials
//   block: one output position p (of 1024) x one batch tile (16 rows).
//   Both col-tiles of a wave share A -> one A load feeds two WMMAs.
// ---------------------------------------------------------------------------
__global__ void __launch_bounds__(256) deconv1_kernel(
    const bf16* __restrict__ xs, const bf16* __restrict__ wd1T,
    const float* __restrict__ b_dc1, bf16* __restrict__ y1,
    float* __restrict__ bnsum, float* __restrict__ bnsq) {
  __shared__ bf16 stage[16 * 256];
  __shared__ float lsum[256];
  __shared__ float lsq[256];
  const int tid = threadIdx.x, wave = tid >> 5, lane = tid & 31;
  const int bid = blockIdx.x;
  const int p = bid >> 2, bt = bid & 3;
  const int m = p >> 1, odd = p & 1;
  lsum[tid] = 0.f; lsq[tid] = 0.f;
  __syncthreads();
  const bf16* A0 = xs + (size_t)m * 16384;
  const int c0 = wave * 32;                    // two 16-wide col tiles
  v8f acc0 = {0.f, 0.f, 0.f, 0.f, 0.f, 0.f, 0.f, 0.f};
  v8f acc1 = {0.f, 0.f, 0.f, 0.f, 0.f, 0.f, 0.f, 0.f};
  if (!odd) {
    const bf16* Bw = wd1T + 65536;             // k=1
    for (int k0 = 0; k0 < 256; k0 += 32) {
      v16bf a = load_a16(A0, 256, bt * 16, k0);
      acc0 = wmma_bf16(a, load_bT16(Bw, 256, c0, k0), acc0);
      acc1 = wmma_bf16(a, load_bT16(Bw, 256, c0 + 16, k0), acc1);
    }
  } else {
    const bf16* B2 = wd1T + 2 * 65536;         // k=2 on x[m]
    for (int k0 = 0; k0 < 256; k0 += 32) {
      v16bf a = load_a16(A0, 256, bt * 16, k0);
      acc0 = wmma_bf16(a, load_bT16(B2, 256, c0, k0), acc0);
      acc1 = wmma_bf16(a, load_bT16(B2, 256, c0 + 16, k0), acc1);
    }
    if (m + 1 < 512) {                         // k=0 on x[m+1]
      const bf16* B0w = wd1T;
      const bf16* A1 = A0 + 16384;
      for (int k0 = 0; k0 < 256; k0 += 32) {
        v16bf a = load_a16(A1, 256, bt * 16, k0);
        acc0 = wmma_bf16(a, load_bT16(B0w, 256, c0, k0), acc0);
        acc1 = wmma_bf16(a, load_bT16(B0w, 256, c0 + 16, k0), acc1);
      }
    }
  }
  const int n = lane & 15;
#pragma unroll
  for (int s = 0; s < 2; ++s) {
    const v8f acc = s ? acc1 : acc0;
    const int co = c0 + s * 16 + n;
    const float bias = b_dc1[co];
    float s1 = 0.f, s2 = 0.f;
#pragma unroll
    for (int v = 0; v < 8; ++v) {
      int rr = ((lane >> 4) << 3) + v;
      float val = acc[v] + bias;
      val = val > 0.f ? val : 0.f;             // ReLU before BN
      stage[rr * 256 + co] = (bf16)val;
      s1 += val; s2 += val * val;
    }
    atomicAdd(&lsum[co], s1);
    atomicAdd(&lsq[co], s2);
  }
  __syncthreads();
  const int rr = tid >> 4, cc = (tid & 15) * 16;
  bf16* dst = y1 + (size_t)p * 16384 + (size_t)(bt * 16 + rr) * 256 + cc;
#pragma unroll
  for (int j = 0; j < 16; ++j) dst[j] = stage[rr * 256 + cc + j];
  atomicAdd(&bnsum[tid], lsum[tid]);
  atomicAdd(&bnsq[tid], lsq[tid]);
}

// ---------------------------------------------------------------------------
// BatchNorm finalize: per-channel scale/shift
// ---------------------------------------------------------------------------
__global__ void bnfin_kernel(const float* __restrict__ bnsum,
                             const float* __restrict__ bnsq,
                             const float* __restrict__ gamma,
                             const float* __restrict__ beta,
                             float* __restrict__ scale, float* __restrict__ shift) {
  const int c = threadIdx.x;
  const float N = 65536.f;                     // 64 batch * 1024 positions
  float mean = bnsum[c] / N;
  float var = bnsq[c] / N - mean * mean;
  float inv = rsqrtf(var + 1e-5f);
  float sc = gamma[c] * inv;
  scale[c] = sc;
  shift[c] = beta[c] - mean * sc;
}

// ---------------------------------------------------------------------------
// Deconv2: BN folded into A-fragment load (shared across both col tiles),
// fp32 output [B, 4T, F]
// ---------------------------------------------------------------------------
__global__ void __launch_bounds__(256) deconv2_kernel(
    const bf16* __restrict__ y1, const bf16* __restrict__ wd2T,
    const float* __restrict__ b_dc2, const float* __restrict__ scale,
    const float* __restrict__ shift, float* __restrict__ out) {
  __shared__ float stage[16 * 256];
  const int tid = threadIdx.x, wave = tid >> 5, lane = tid & 31;
  const int bid = blockIdx.x;
  const int j = bid >> 2, bt = bid & 3;
  const int m = j >> 1, odd = j & 1;
  const bf16* A0 = y1 + (size_t)m * 16384;
  const int c0 = wave * 32;
  v8f acc0 = {0.f, 0.f, 0.f, 0.f, 0.f, 0.f, 0.f, 0.f};
  v8f acc1 = {0.f, 0.f, 0.f, 0.f, 0.f, 0.f, 0.f, 0.f};
  if (!odd) {
    const bf16* Bw = wd2T + 65536;             // k=1
    for (int k0 = 0; k0 < 256; k0 += 32) {
      v16bf a = load_a16_bn(A0, 256, bt * 16, k0, scale, shift);
      acc0 = wmma_bf16(a, load_bT16(Bw, 256, c0, k0), acc0);
      acc1 = wmma_bf16(a, load_bT16(Bw, 256, c0 + 16, k0), acc1);
    }
  } else {
    const bf16* B2 = wd2T + 2 * 65536;         // k=2 on y[m]
    for (int k0 = 0; k0 < 256; k0 += 32) {
      v16bf a = load_a16_bn(A0, 256, bt * 16, k0, scale, shift);
      acc0 = wmma_bf16(a, load_bT16(B2, 256, c0, k0), acc0);
      acc1 = wmma_bf16(a, load_bT16(B2, 256, c0 + 16, k0), acc1);
    }
    if (m + 1 < 1024) {                        // k=0 on y[m+1]
      const bf16* B0w = wd2T;
      const bf16* A1 = A0 + 16384;
      for (int k0 = 0; k0 < 256; k0 += 32) {
        v16bf a = load_a16_bn(A1, 256, bt * 16, k0, scale, shift);
        acc0 = wmma_bf16(a, load_bT16(B0w, 256, c0, k0), acc0);
        acc1 = wmma_bf16(a, load_bT16(B0w, 256, c0 + 16, k0), acc1);
      }
    }
  }
  const int n = lane & 15;
#pragma unroll
  for (int s = 0; s < 2; ++s) {
    const v8f acc = s ? acc1 : acc0;
    const int co = c0 + s * 16 + n;
    const float bias = b_dc2[co];
#pragma unroll
    for (int v = 0; v < 8; ++v) {
      int rr = ((lane >> 4) << 3) + v;
      stage[rr * 256 + co] = acc[v] + bias;
    }
  }
  __syncthreads();
  const int rr = tid >> 4, cc = (tid & 15) * 16;
  const int b = bt * 16 + rr;
  float* dst = out + (size_t)b * (2048 * 256) + (size_t)j * 256 + cc;
#pragma unroll
  for (int q = 0; q < 16; ++q) dst[q] = stage[rr * 256 + cc + q];
}

// ---------------------------------------------------------------------------
// Host-side launch
// ---------------------------------------------------------------------------
extern "C" void kernel_launch(void* const* d_in, const int* in_sizes, int n_in,
                              void* d_out, int out_size, void* d_ws, size_t ws_size,
                              hipStream_t stream) {
  (void)in_sizes; (void)n_in; (void)out_size; (void)ws_size;
  const float* h0      = (const float*)d_in[0];
  const float* c0      = (const float*)d_in[1];
  /* d_in[2] = seq_len (fixed 512) */
  const float* W_ih    = (const float*)d_in[3];
  const float* b_ih    = (const float*)d_in[4];
  const float* W_hh    = (const float*)d_in[5];
  const float* b_hh    = (const float*)d_in[6];
  const float* W_dense = (const float*)d_in[7];
  const float* b_dense = (const float*)d_in[8];
  const float* w_dc1   = (const float*)d_in[9];
  const float* b_dc1   = (const float*)d_in[10];
  const float* w_dc2   = (const float*)d_in[11];
  const float* b_dc2   = (const float*)d_in[12];
  const float* gamma   = (const float*)d_in[13];
  const float* beta    = (const float*)d_in[14];
  float* out = (float*)d_out;

  char* ws = (char*)d_ws;
  // workspace layout (bytes)
  unsigned* sync_cnt = (unsigned*)(ws + 0);                   // 256 reserved
  bf16* hA    = (bf16*)(ws + 256);                            // 64*512 bf16
  bf16* hB    = hA + 64 * 512;                                // 64*512 bf16
  bf16* wd1T  = (bf16*)(ws + 256 + 131072);                   // 3*256*256 bf16
  bf16* wd2T  = wd1T + 3 * 65536;                             // 3*256*256 bf16
  float* bnsum = (float*)(ws + 256 + 131072 + 786432);        // 256 f32
  float* bnsq  = bnsum + 256;
  float* scale = bnsq + 256;
  float* shift = scale + 256;
  bf16* xbuf  = (bf16*)((char*)(shift + 256) + 224);          // align; 513*64*256 bf16
  bf16* y1    = xbuf + (size_t)513 * 16384;                   // 1024*64*256 bf16

  // 1) prep: conversions, transposes, zero counters
  prep_kernel<<<1666, 256, 0, stream>>>(h0, w_dc1, w_dc2, hA, wd1T, wd2T,
                                        bnsum, bnsq, sync_cnt);

  // 2) persistent LSTM scan (32 WGs, 132 KB LDS each)
  const size_t lstm_lds = (size_t)(64 * 256 + 64 * 512 + 16 * 512) * 2 +
                          (size_t)(64 * 64 + 64 * 16 + 64 + 16) * 4;
  lstm_kernel<<<NWG_LSTM, 256, lstm_lds, stream>>>(
      W_ih, b_ih, W_hh, b_hh, W_dense, b_dense, c0, hA, hB, xbuf, sync_cnt);

  // 3) deconv1 + relu + BN partial stats   (xs = xbuf[1..512])
  deconv1_kernel<<<4096, 256, 0, stream>>>(xbuf + 16384, wd1T, b_dc1, y1,
                                           bnsum, bnsq);

  // 4) BN finalize
  bnfin_kernel<<<1, 256, 0, stream>>>(bnsum, bnsq, gamma, beta, scale, shift);

  // 5) deconv2 with BN folded into A loads, fp32 output
  deconv2_kernel<<<8192, 256, 0, stream>>>(y1, wd2T, b_dc2, scale, shift, out);
}
